// AttentionBlock_15281493639647
// MI455X (gfx1250) — matrix operand order repaired and checked by reading, not compiled, not run on hardware
//
#include <hip/hip_runtime.h>
#include <hip/hip_bf16.h>

typedef __attribute__((ext_vector_type(16))) __bf16 v16bf;
typedef __attribute__((ext_vector_type(8)))  __bf16 v8bf;
typedef __attribute__((ext_vector_type(8)))  float  v8f;

#define Bc 16
#define Cc 512
#define Lc 1024
#define Hc 8
#define CHc 64
#define Gc 32
#define GCH 16   // channels per group

#define WMMA_BF16(a, b, c) \
  __builtin_amdgcn_wmma_f32_16x16x32_bf16(false, (a), false, (b), (short)0, (c), false, false)

// A-operand (16x32, MxK) loader. base -> row 0 / K 0, rows strided by ld.
// Per ISA: lanes 0-15 hold row m=lane, K {0..7, 16..23}; lanes 16-31 K {8..15, 24..31}.
static __device__ __forceinline__ v16bf load_tA(const __bf16* base, int ld) {
  const int lane = threadIdx.x & 31;
  const int m    = lane & 15;
  const int off  = (lane >> 4) << 3;
  const __bf16* p = base + (size_t)m * ld + off;
  union { v16bf v; v8bf h[2]; } u;
  u.h[0] = *(const v8bf*)(p);
  u.h[1] = *(const v8bf*)(p + 16);
  return u.v;
}

// B-operand (32x16, KxN) loader, memory layout column-major in K (each column contiguous).
// Per ISA: lanes 0-15 hold col n=lane, K {0..7, 16..23}; lanes 16-31 K {8..15, 24..31}.
static __device__ __forceinline__ v16bf load_tB(const __bf16* base, int ld) {
  const int lane = threadIdx.x & 31;
  const int n    = lane & 15;
  const int koff = (lane >> 4) << 3;
  const __bf16* p = base + (size_t)n * ld + koff;
  union { v16bf v; v8bf h[2]; } u;
  u.h[0] = *(const v8bf*)(p);
  u.h[1] = *(const v8bf*)(p + 16);
  return u.v;
}

// Async 16B global->LDS copy (CDNA5 GLOBAL_LOAD_ASYNC_TO_LDS_B128, ASYNCcnt tracked).
static __device__ __forceinline__ void async_copy_b128(unsigned lds_addr, const void* gptr) {
  asm volatile("global_load_async_to_lds_b128 %0, %1, off"
               :: "v"(lds_addr), "v"(gptr) : "memory");
}
static __device__ __forceinline__ void wait_async0() {
  asm volatile("s_wait_asynccnt 0x0" ::: "memory");
}

__global__ void __launch_bounds__(256) cvt_kernel(const float* __restrict__ src,
                                                  __bf16* __restrict__ dst, int nel) {
  int i = blockIdx.x * 256 + threadIdx.x;
  if (i < nel) dst[i] = (__bf16)src[i];
}

// GroupNorm over (b,g): 16 channels x 1024 positions (contiguous block).
// Writes hid transposed to (b, L, C) bf16.
__global__ void __launch_bounds__(256) gn_kernel(const float* __restrict__ x,
                                                 const float* __restrict__ w,
                                                 const float* __restrict__ bi,
                                                 __bf16* __restrict__ hid) {
  const int b = blockIdx.x >> 5;
  const int g = blockIdx.x & 31;
  const float* xg = x + ((size_t)b * Cc + (size_t)g * GCH) * Lc;
  float s = 0.f, s2 = 0.f;
  for (int i = threadIdx.x; i < GCH * Lc; i += 256) {
    float vv = xg[i]; s += vv; s2 += vv * vv;
  }
  for (int d = 1; d < 32; d <<= 1) { s += __shfl_xor(s, d, 32); s2 += __shfl_xor(s2, d, 32); }
  __shared__ float r0[8], r1[8];
  __shared__ float mu_s, rs_s;
  const int wv = threadIdx.x >> 5;
  if ((threadIdx.x & 31) == 0) { r0[wv] = s; r1[wv] = s2; }
  __syncthreads();
  if (threadIdx.x == 0) {
    float ts = 0.f, ts2 = 0.f;
    for (int i = 0; i < 8; ++i) { ts += r0[i]; ts2 += r1[i]; }
    const float inv = 1.0f / (float)(GCH * Lc);
    float mu = ts * inv;
    float var = ts2 * inv - mu * mu;
    mu_s = mu;
    rs_s = rsqrtf(var + 1e-5f);
  }
  __syncthreads();
  const float mu = mu_s, rs = rs_s;
  for (int i = threadIdx.x; i < GCH * Lc; i += 256) {
    const int cl = i & 15;
    const int l  = i >> 4;
    const int c  = g * GCH + cl;
    float vv = x[((size_t)b * Cc + c) * Lc + l];
    float o  = (vv - mu) * rs * w[c] + bi[c];
    hid[((size_t)b * Lc + l) * Cc + c] = (__bf16)o;
  }
}

// QKV GEMM: qkv[o,l] = sum_c W[o,c]*hid[b,l,c]. Block tile 128(M) x 64(N), 8 waves.
// Double-buffered async global->LDS pipeline: DMA for tile i+1 overlaps WMMAs on tile i.
__global__ void __launch_bounds__(256) qkv_kernel(const __bf16* __restrict__ wq,
                                                  const __bf16* __restrict__ hid,
                                                  const float* __restrict__ bias,
                                                  __bf16* __restrict__ q,
                                                  __bf16* __restrict__ k,
                                                  __bf16* __restrict__ v) {
  __shared__ __bf16 Bs[2][64 * 40];
  const int b    = blockIdx.z;
  const int m0   = blockIdx.x * 128;
  const int n0   = blockIdx.y * 64;
  const int wave = threadIdx.x >> 5;
  const int lane = threadIdx.x & 31;
  const __bf16* hb = hid + (size_t)b * Lc * Cc;
  v8f acc0 = {}, acc1 = {}, acc2 = {}, acc3 = {};
  const int nrow = threadIdx.x >> 2;
  const int seg  = threadIdx.x & 3;
  const unsigned lds0 = (unsigned)(size_t)&Bs[0][nrow * 40 + seg * 8];
  const unsigned lds1 = (unsigned)(size_t)&Bs[1][nrow * 40 + seg * 8];
  const __bf16* gsrc = &hb[(size_t)(n0 + nrow) * Cc + seg * 8];
  async_copy_b128(lds0, gsrc);             // prologue: tile 0 in flight
  #pragma unroll 2
  for (int k0 = 0; k0 < Cc; k0 += 32) {
    const int buf = (k0 >> 5) & 1;
    wait_async0();                         // current tile landed (this wave)
    __syncthreads();                       // ...for all waves; prior readers of other buf retired
    if (k0 + 32 < Cc)
      async_copy_b128(buf ? lds0 : lds1, gsrc + k0 + 32);  // next tile behind the WMMAs
    const __bf16* Bb = &Bs[buf][0];
    v16bf a = load_tA(wq + (size_t)(m0 + wave * 16) * Cc + k0, Cc);
    acc0 = WMMA_BF16(a, load_tB(&Bb[ 0 * 40], 40), acc0);
    acc1 = WMMA_BF16(a, load_tB(&Bb[16 * 40], 40), acc1);
    acc2 = WMMA_BF16(a, load_tB(&Bb[32 * 40], 40), acc2);
    acc3 = WMMA_BF16(a, load_tB(&Bb[48 * 40], 40), acc3);
    __syncthreads();                       // this buf's readers done before it is overwritten
  }
  const int nl = lane & 15;
  const int mh = (lane >> 4) * 8;
  const float scale = 0.35355339059327373f;   // 64^-0.25
  v8f accs[4] = {acc0, acc1, acc2, acc3};
  for (int nt = 0; nt < 4; ++nt) {
    const int l = n0 + nt * 16 + nl;
    for (int r = 0; r < 8; ++r) {
      const int o = m0 + wave * 16 + mh + r;
      float val = accs[nt][r] + bias[o];
      const int which = o >> 9;   // uniform per wave
      const int oc = o & 511;
      const int h = oc >> 6;
      const int c = oc & 63;
      if (which == 0)      q[(((size_t)b * Hc + h) * Lc + l) * CHc + c] = (__bf16)(val * scale);
      else if (which == 1) k[(((size_t)b * Hc + h) * Lc + l) * CHc + c] = (__bf16)(val * scale);
      else                 v[(((size_t)b * Hc + h) * CHc + c) * Lc + l] = (__bf16)val;
    }
  }
}

// Flash attention: each wave owns 16 t-rows, streams s in 32-chunks, online softmax.
__global__ void __launch_bounds__(256) attn_kernel(const __bf16* __restrict__ q,
                                                   const __bf16* __restrict__ k,
                                                   const __bf16* __restrict__ v,
                                                   __bf16* __restrict__ aout) {
  __shared__ __bf16 Ps[8][16 * 40];   // per-wave P tile, padded stride
  const int bh   = blockIdx.x >> 3;
  const int tg   = blockIdx.x & 7;
  const int wave = threadIdx.x >> 5;
  const int lane = threadIdx.x & 31;
  const int b = bh >> 3;
  const int h = bh & 7;
  const int t0 = tg * 128 + wave * 16;
  const __bf16* qb = q + (size_t)bh * Lc * CHc;
  const __bf16* kb = k + (size_t)bh * Lc * CHc;
  const __bf16* vb = v + (size_t)bh * CHc * Lc;
  const v16bf a0 = load_tA(qb + (size_t)t0 * CHc +  0, CHc);
  const v16bf a1 = load_tA(qb + (size_t)t0 * CHc + 32, CHc);
  v8f acc0 = {}, acc1 = {}, acc2 = {}, acc3 = {};
  float mi[8], li[8];
  for (int r = 0; r < 8; ++r) { mi[r] = -1e30f; li[r] = 0.f; }
  __bf16* myP = &Ps[wave][0];
  const float LOG2E = 1.4426950408889634f;
  const int nl   = lane & 15;
  const int mrow = (lane >> 4) * 8;
  for (int s0 = 0; s0 < Lc; s0 += 32) {
    if (s0 + 32 < Lc) {   // GL2 prefetch of next chunk's K rows / V columns
      __builtin_prefetch(kb + (size_t)(s0 + 32 + (lane & 31)) * CHc, 0, 3);
      __builtin_prefetch(vb + (size_t)(lane & 31) * Lc + s0 + 32, 0, 3);
    }
    v8f z0 = {}, z1 = {};
    v8f S0 = WMMA_BF16(a1, load_tB(kb + (size_t)s0 * CHc + 32, CHc),
             WMMA_BF16(a0, load_tB(kb + (size_t)s0 * CHc +  0, CHc), z0));
    v8f S1 = WMMA_BF16(a1, load_tB(kb + (size_t)(s0 + 16) * CHc + 32, CHc),
             WMMA_BF16(a0, load_tB(kb + (size_t)(s0 + 16) * CHc +  0, CHc), z1));
    float mnew[8], corr[8];
    for (int r = 0; r < 8; ++r) {
      float mx = fmaxf(S0[r], S1[r]);
      mx = fmaxf(mx, __shfl_xor(mx, 1, 32));
      mx = fmaxf(mx, __shfl_xor(mx, 2, 32));
      mx = fmaxf(mx, __shfl_xor(mx, 4, 32));
      mx = fmaxf(mx, __shfl_xor(mx, 8, 32));
      mnew[r] = fmaxf(mi[r], mx);
      corr[r] = __builtin_amdgcn_exp2f((mi[r] - mnew[r]) * LOG2E);
    }
    float rsum[8];
    for (int r = 0; r < 8; ++r) {
      float p0 = __builtin_amdgcn_exp2f((S0[r] - mnew[r]) * LOG2E);
      float p1 = __builtin_amdgcn_exp2f((S1[r] - mnew[r]) * LOG2E);
      rsum[r] = p0 + p1;
      myP[(mrow + r) * 40 + nl]      = (__bf16)p0;
      myP[(mrow + r) * 40 + 16 + nl] = (__bf16)p1;
    }
    for (int r = 0; r < 8; ++r) {
      float rs = rsum[r];
      rs += __shfl_xor(rs, 1, 32);
      rs += __shfl_xor(rs, 2, 32);
      rs += __shfl_xor(rs, 4, 32);
      rs += __shfl_xor(rs, 8, 32);
      li[r] = li[r] * corr[r] + rs;
      mi[r] = mnew[r];
    }
    for (int r = 0; r < 8; ++r) {
      acc0[r] *= corr[r]; acc1[r] *= corr[r]; acc2[r] *= corr[r]; acc3[r] *= corr[r];
    }
    // P written by this wave only; DS ops are in-order per wave after the wait.
    asm volatile("s_wait_dscnt 0x0" ::: "memory");
    const v16bf pa = load_tA(myP, 40);
    acc0 = WMMA_BF16(pa, load_tB(vb + (size_t) 0 * Lc + s0, Lc), acc0);
    acc1 = WMMA_BF16(pa, load_tB(vb + (size_t)16 * Lc + s0, Lc), acc1);
    acc2 = WMMA_BF16(pa, load_tB(vb + (size_t)32 * Lc + s0, Lc), acc2);
    acc3 = WMMA_BF16(pa, load_tB(vb + (size_t)48 * Lc + s0, Lc), acc3);
  }
  v8f accs[4] = {acc0, acc1, acc2, acc3};
  for (int nt = 0; nt < 4; ++nt) {
    const int c = h * CHc + nt * 16 + nl;
    for (int r = 0; r < 8; ++r) {
      const int t = t0 + mrow + r;
      float o = accs[nt][r] / li[r];
      aout[((size_t)b * Lc + t) * Cc + c] = (__bf16)o;
    }
  }
}

// Proj GEMM + residual: out[b,o,l] = x[b,o,l] + sum_c Wp[o,c]*a[b,l,c] + bias[o]
__global__ void __launch_bounds__(256) proj_kernel(const __bf16* __restrict__ wp,
                                                   const __bf16* __restrict__ ain,
                                                   const float* __restrict__ bias,
                                                   const float* __restrict__ x,
                                                   float* __restrict__ out) {
  __shared__ __bf16 Bs[2][64 * 40];
  const int b    = blockIdx.z;
  const int m0   = blockIdx.x * 128;
  const int n0   = blockIdx.y * 64;
  const int wave = threadIdx.x >> 5;
  const int lane = threadIdx.x & 31;
  const __bf16* ab = ain + (size_t)b * Lc * Cc;
  v8f acc0 = {}, acc1 = {}, acc2 = {}, acc3 = {};
  const int nrow = threadIdx.x >> 2;
  const int seg  = threadIdx.x & 3;
  const unsigned lds0 = (unsigned)(size_t)&Bs[0][nrow * 40 + seg * 8];
  const unsigned lds1 = (unsigned)(size_t)&Bs[1][nrow * 40 + seg * 8];
  const __bf16* gsrc = &ab[(size_t)(n0 + nrow) * Cc + seg * 8];
  async_copy_b128(lds0, gsrc);
  #pragma unroll 2
  for (int k0 = 0; k0 < Cc; k0 += 32) {
    const int buf = (k0 >> 5) & 1;
    wait_async0();
    __syncthreads();
    if (k0 + 32 < Cc)
      async_copy_b128(buf ? lds0 : lds1, gsrc + k0 + 32);
    const __bf16* Bb = &Bs[buf][0];
    v16bf a = load_tA(wp + (size_t)(m0 + wave * 16) * Cc + k0, Cc);
    acc0 = WMMA_BF16(a, load_tB(&Bb[ 0 * 40], 40), acc0);
    acc1 = WMMA_BF16(a, load_tB(&Bb[16 * 40], 40), acc1);
    acc2 = WMMA_BF16(a, load_tB(&Bb[32 * 40], 40), acc2);
    acc3 = WMMA_BF16(a, load_tB(&Bb[48 * 40], 40), acc3);
    __syncthreads();
  }
  const int nl = lane & 15;
  const int mh = (lane >> 4) * 8;
  v8f accs[4] = {acc0, acc1, acc2, acc3};
  for (int nt = 0; nt < 4; ++nt) {
    const int l = n0 + nt * 16 + nl;
    for (int r = 0; r < 8; ++r) {
      const int o = m0 + wave * 16 + mh + r;
      const size_t idx = ((size_t)b * Cc + o) * Lc + l;
      out[idx] = x[idx] + accs[nt][r] + bias[o];
    }
  }
}

extern "C" void kernel_launch(void* const* d_in, const int* in_sizes, int n_in,
                              void* d_out, int out_size, void* d_ws, size_t ws_size,
                              hipStream_t stream) {
  const float* x      = (const float*)d_in[0];
  const float* norm_w = (const float*)d_in[1];
  const float* norm_b = (const float*)d_in[2];
  const float* qkv_w  = (const float*)d_in[3];
  const float* qkv_b  = (const float*)d_in[4];
  const float* proj_w = (const float*)d_in[5];
  const float* proj_b = (const float*)d_in[6];
  float* out = (float*)d_out;

  char* ws = (char*)d_ws;
  const size_t MB = 1ull << 20;
  __bf16* hid = (__bf16*)(ws +  0 * MB);   // (B,L,C)   16 MB
  __bf16* wqb = (__bf16*)(ws + 16 * MB);   // 1536x512  1.5 MB
  __bf16* wpb = (__bf16*)(ws + 18 * MB);   // 512x512   0.5 MB
  __bf16* qb  = (__bf16*)(ws + 19 * MB);   // (B,H,L,ch) 16 MB
  __bf16* kb  = (__bf16*)(ws + 35 * MB);   // (B,H,L,ch) 16 MB
  __bf16* vb  = (__bf16*)(ws + 51 * MB);   // (B,H,ch,L) 16 MB
  __bf16* ao  = (__bf16*)(ws + 67 * MB);   // (B,L,C)   16 MB

  cvt_kernel<<<(1536 * 512) / 256, 256, 0, stream>>>(qkv_w, wqb, 1536 * 512);
  cvt_kernel<<<(512 * 512) / 256, 256, 0, stream>>>(proj_w, wpb, 512 * 512);
  gn_kernel<<<Bc * Gc, 256, 0, stream>>>(x, norm_w, norm_b, hid);
  qkv_kernel<<<dim3(12, 16, Bc), 256, 0, stream>>>(wqb, hid, qkv_b, qb, kb, vb);
  attn_kernel<<<Bc * Hc * 8, 256, 0, stream>>>(qb, kb, vb, ao);
  proj_kernel<<<dim3(4, 16, Bc), 256, 0, stream>>>(wpb, ao, proj_b, x, out);
}
